// ConditionalRandomField_30777735643367
// MI455X (gfx1250) — matrix operand order repaired and checked by reading, not compiled, not run on hardware
//
#include <hip/hip_runtime.h>
#include <hip/hip_bf16.h>

typedef __attribute__((ext_vector_type(16))) __bf16 v16bf;
typedef __attribute__((ext_vector_type(8)))  float  v8f;

#define K_TAGS   64
#define BOS_TAG  63
#define EOS_TAG  62
#define D_DIM    128
#define T_LEN    256
#define N_BATCH  2048
#define V_VOCAB  50000
#define TINY_F   1e-45f
#define LN2_F    0.69314718055994530942f

// ---------------------------------------------------------------------------
// Phase 1: emisT[w][tag] = exp(dot(E[w], ThetaB[tag])), tags 62/63 forced TINY.
// One wave computes a 16-word x 64-tag tile via 16 bf16 WMMAs (K=128).
// ---------------------------------------------------------------------------
__global__ __launch_bounds__(128) void crf_emis_gemm(
    const float* __restrict__ ThetaB,   // [64][128]
    const float* __restrict__ E,        // [V+2][128]
    float* __restrict__ emisT)          // [V][64]
{
  const int lane = threadIdx.x & 31;
  const int wave = threadIdx.x >> 5;
  const int lrow = lane & 15;
  const int half = lane >> 4;
  const int w0   = (blockIdx.x * 4 + wave) * 16;

  int myrow = w0 + lrow;
  if (myrow >= V_VOCAB) myrow = V_VOCAB - 1;       // clamp for safe loads
  const float* erow = E + (size_t)myrow * D_DIM;

  v8f acc[4];
#pragma unroll
  for (int n = 0; n < 4; ++n) { v8f z = {}; acc[n] = z; }

#pragma unroll
  for (int kk = 0; kk < 4; ++kk) {
    // A-operand: 16x32 bf16, lane half interleave per ISA layout table.
    v16bf a;
#pragma unroll
    for (int v = 0; v < 8; ++v) {
      int kb = 32*kk + ((v & 4) ? 16 : 0) + 8*half + 2*(v & 3);
      a[2*v]   = (__bf16)erow[kb];
      a[2*v+1] = (__bf16)erow[kb + 1];
    }
#pragma unroll
    for (int n = 0; n < 4; ++n) {
      // B-operand: ThetaB^T (K=feature, N=tag), N = lane%16 per column tile.
      const float* trow = ThetaB + (size_t)(16*n + lrow) * D_DIM;
      v16bf b;
#pragma unroll
      for (int v = 0; v < 8; ++v) {
        int kb = 32*kk + ((v & 4) ? 16 : 0) + 8*half + 2*(v & 3);
        b[2*v]   = (__bf16)trow[kb];
        b[2*v+1] = (__bf16)trow[kb + 1];
      }
      acc[n] = __builtin_amdgcn_wmma_f32_16x16x32_bf16(
          false, a, false, b, (short)0, acc[n], false, false);
    }
  }

  // exp, TINY override for EOS/BOS tag rows, word-major store (coalesced 64B).
#pragma unroll
  for (int n = 0; n < 4; ++n) {
    int tag = 16*n + lrow;
#pragma unroll
    for (int j = 0; j < 8; ++j) {
      int w = w0 + j + 8*half;
      if (w < V_VOCAB) {
        float val = __expf(acc[n][j]);
        if (tag >= EOS_TAG) val = TINY_F;
        emisT[(size_t)w * K_TAGS + tag] = val;
      }
    }
  }
}

// ---------------------------------------------------------------------------
// Phase 2: forward scan. One wave owns 16 sentences; alpha lives in WMMA C
// tiles, A (=exp(WA), col BOS zeroed) lives in registers as bf16 B-operands.
// Per step: LDS transpose + row-max rescale -> 8 WMMAs -> * gathered emis.
// Emissions for t+1 are prefetched (register double buffer).
// ---------------------------------------------------------------------------
__global__ __launch_bounds__(128) void crf_forward_scan(
    const int*   __restrict__ words,    // [BATCH][T]
    const float* __restrict__ WA,       // [64][64]
    const float* __restrict__ emisT,    // [V][64]
    float* __restrict__ out)            // [BATCH]
{
  __shared__ __align__(16) float sA[K_TAGS * K_TAGS];   // exp(WA), col 63 = 0
  __shared__ __align__(16) float sAlpha[4][16 * 68];    // per-wave, padded rows

  const int tid  = threadIdx.x;
  const int lane = tid & 31;
  const int wave = tid >> 5;
  const int lrow = lane & 15;
  const int half = lane >> 4;

  for (int i = tid; i < K_TAGS * K_TAGS; i += 128) {
    float v = __expf(WA[i]);
    if ((i & 63) == BOS_TAG) v = 0.0f;
    sA[i] = v;
  }
  __syncthreads();

  // Transition matrix as 8 resident bf16 B-operands: [tile n][K-half kk].
  v16bf Btr[4][2];
#pragma unroll
  for (int n = 0; n < 4; ++n)
#pragma unroll
    for (int kk = 0; kk < 2; ++kk) {
      v16bf b;
#pragma unroll
      for (int v = 0; v < 8; ++v) {
        int k0 = 32*kk + ((v & 4) ? 16 : 0) + 8*half + 2*(v & 3);
        b[2*v]   = (__bf16)sA[(k0    ) * K_TAGS + 16*n + lrow];
        b[2*v+1] = (__bf16)sA[(k0 + 1) * K_TAGS + 16*n + lrow];
      }
      Btr[n][kk] = b;
    }

  const int w0 = (blockIdx.x * 4 + wave) * 16;       // first sentence of wave
  float* aRow = &sAlpha[wave][0];
  const float* rp = aRow + lrow * 68;                // my sentence's row

  // alpha0: one-hot at BOS tag (N=63 -> tile 3, lane%16==15), every sentence.
  v8f acc[4];
#pragma unroll
  for (int n = 0; n < 4; ++n) { v8f z = {}; acc[n] = z; }
  if (lrow == 15) {
#pragma unroll
    for (int j = 0; j < 8; ++j) acc[3][j] = 1.0f;
  }

  float logAcc = 0.0f;                 // sum of log2(scale) for sentence lrow
  float eA[4][8], eB[4][8];            // emission double buffer (C layout)

  auto gather = [&](int t, float (*e)[8]) {
#pragma unroll
    for (int j = 0; j < 8; ++j) {
      int M = j + 8*half;                                    // sentence slot
      int w = words[(size_t)(w0 + M) * T_LEN + t];
      const float* rowp = emisT + (size_t)w * K_TAGS;
#pragma unroll
      for (int n = 0; n < 4; ++n) e[n][j] = rowp[16*n + lrow];
    }
  };

  gather(1, eA);

#pragma unroll 2
  for (int t = 1; t <= T_LEN - 2; ++t) {
    float (*ec)[8] = (t & 1) ? eA : eB;
    float (*en)[8] = (t & 1) ? eB : eA;
    if (t < T_LEN - 2) gather(t + 1, en);      // prefetch next step's emis

    // alpha C tiles -> LDS (bank-conflict-free via 68-float row stride)
#pragma unroll
    for (int n = 0; n < 4; ++n)
#pragma unroll
      for (int j = 0; j < 8; ++j)
        aRow[(j + 8*half) * 68 + 16*n + lrow] = acc[n][j];
    __builtin_amdgcn_wave_barrier();           // keep cross-lane LDS ordering

    // per-sentence max (both halves compute row lrow redundantly)
    float mx = 0.0f;
#pragma unroll
    for (int i = 0; i < 16; ++i) {
      float4 q = ((const float4*)rp)[i];
      mx = fmaxf(mx, fmaxf(fmaxf(q.x, q.y), fmaxf(q.z, q.w)));
    }
    float inv = 1.0f / mx;
    logAcc += __log2f(mx);

    // rescaled alpha -> bf16 A-operands (ISA 16x32 half-lane layout)
    v16bf aop[2];
#pragma unroll
    for (int kk = 0; kk < 2; ++kk) {
      v16bf a;
#pragma unroll
      for (int v = 0; v < 8; ++v) {
        int kb = 32*kk + ((v & 4) ? 16 : 0) + 8*half + 2*(v & 3);
        a[2*v]   = (__bf16)(rp[kb]     * inv);
        a[2*v+1] = (__bf16)(rp[kb + 1] * inv);
      }
      aop[kk] = a;
    }

    // alpha_new = (alpha/scale) @ A  * emis_t     (8 WMMAs, f32 accumulate)
#pragma unroll
    for (int n = 0; n < 4; ++n) {
      v8f c = {};
      c = __builtin_amdgcn_wmma_f32_16x16x32_bf16(false, aop[0], false, Btr[n][0],
                                                  (short)0, c, false, false);
      c = __builtin_amdgcn_wmma_f32_16x16x32_bf16(false, aop[1], false, Btr[n][1],
                                                  (short)0, c, false, false);
#pragma unroll
      for (int j = 0; j < 8; ++j) acc[n][j] = c[j] * ec[n][j];
    }
  }

  // final transition into EOS tag: z = (alpha/scale) . A[:, EOS]
#pragma unroll
  for (int n = 0; n < 4; ++n)
#pragma unroll
    for (int j = 0; j < 8; ++j)
      aRow[(j + 8*half) * 68 + 16*n + lrow] = acc[n][j];
  __builtin_amdgcn_wave_barrier();

  float mx = 0.0f;
#pragma unroll
  for (int i = 0; i < 16; ++i) {
    float4 q = ((const float4*)rp)[i];
    mx = fmaxf(mx, fmaxf(fmaxf(q.x, q.y), fmaxf(q.z, q.w)));
  }
  float z = 0.0f;
#pragma unroll
  for (int k = 0; k < K_TAGS; ++k) z += rp[k] * sA[k * K_TAGS + EOS_TAG];
  z /= mx;

  float logZ = LN2_F * (__log2f(z) + __log2f(mx) + logAcc);
  if (lane < 16) out[w0 + lrow] = logZ;
}

// ---------------------------------------------------------------------------
extern "C" void kernel_launch(void* const* d_in, const int* in_sizes, int n_in,
                              void* d_out, int out_size, void* d_ws, size_t ws_size,
                              hipStream_t stream) {
  const int*   words  = (const int*)  d_in[0];
  const float* ThetaB = (const float*)d_in[1];
  const float* WA     = (const float*)d_in[2];
  const float* E      = (const float*)d_in[3];
  float* out   = (float*)d_out;
  float* emisT = (float*)d_ws;        // V * 64 floats = 12.8 MB scratch

  (void)in_sizes; (void)n_in; (void)out_size; (void)ws_size;

  int gemmBlocks = (V_VOCAB + 63) / 64;          // 64 words per 128-thread WG
  crf_emis_gemm<<<gemmBlocks, 128, 0, stream>>>(ThetaB, E, emisT);
  crf_forward_scan<<<N_BATCH / 64, 128, 0, stream>>>(words, WA, emisT, out);
}
